// sentence_rep_17721035063596
// MI455X (gfx1250) — compile-verified, matrix operand order checked by reading, not run on hardware
//
#include <hip/hip_runtime.h>
#include <hip/hip_bf16.h>

typedef __attribute__((ext_vector_type(16))) _Float16 v16h;
typedef __attribute__((ext_vector_type(8)))  _Float16 v8h;
typedef __attribute__((ext_vector_type(8)))  float    v8f;

#define NWORDS   16384
#define LSEQ     16
#define CDIM     64
#define HDIM     128
#define IDIM     896       // 868 padded to 28*32
#define IEFF     868
#define XSTR     72        // 64 + 8 halfs pad
#define HSTR     136       // 128 + 8 halfs pad
#define ASTR     904       // 896 + 8 halfs pad

// ---- workspace layout (in _Float16 elements) ----
#define OFF_WIHF 0u
#define OFF_WIHR 32768u
#define OFF_WHHF 65536u
#define OFF_WHHR 131072u
#define OFF_WLIN 196608u
#define OFF_INFO 655360u   // total = 655360 + 16384*896 halfs ~= 30.7 MB

__device__ inline v16h mk16(v8h lo, v8h hi) {
  v16h r;
#pragma unroll
  for (int j = 0; j < 8; ++j) { r[j] = lo[j]; r[j + 8] = hi[j]; }
  return r;
}

// A-fragment (16x32 f16, row-major buffer with `stride` halfs): lane m+16s holds
// row m, halfs j<8 -> K = kbase+8s+j ; j>=8 -> K = kbase+16+8s+(j-8)
__device__ inline v16h loadA(const _Float16* buf, int stride, int kbase, int m, int s) {
  const _Float16* p = buf + m * stride + kbase + 8 * s;
  v8h lo = *(const v8h*)p;
  v8h hi = *(const v8h*)(p + 16);
  return mk16(lo, hi);
}

// B-fragment pre-swizzled in global memory: 32 lanes x 16 contiguous halfs
__device__ inline v16h loadB(const _Float16* wp) {
  v8h lo = *(const v8h*)wp;
  v8h hi = *(const v8h*)(wp + 8);
  return mk16(lo, hi);
}

__device__ inline v8f wmma16(v16h a, v16h b, v8f c) {
  return __builtin_amdgcn_wmma_f32_16x16x32_f16(false, a, false, b, (short)0, c,
                                                false, false);
}

// gfx1250 hardware tanh (TRANS32 op, co-executes with WMMA on the XDL pipe)
__device__ inline float fast_tanh(float x) {
#if __has_builtin(__builtin_amdgcn_tanhf)
  return __builtin_amdgcn_tanhf(x);
#else
  float y;
  asm volatile("v_tanh_f32 %0, %1" : "=v"(y) : "v"(x));
  return y;
#endif
}

// sigmoid via tanh identity: no exp / ldexp / exec-mask branches
__device__ inline float sigm(float x) {
  return __builtin_fmaf(0.5f, fast_tanh(0.5f * x), 0.5f);
}

// ---------------------------------------------------------------------------
// Pre-swizzle fp32 weight [R x Keff] (row-major) into per-lane B fragments.
// grid = (R/16, K/32), block = 512 (lane = tid>>4, j = tid&15)
// ---------------------------------------------------------------------------
__global__ __launch_bounds__(512) void swz_kernel(const float* __restrict__ W,
                                                  _Float16* __restrict__ out,
                                                  int Keff) {
  const int t = blockIdx.x, kt = blockIdx.y, nkt = gridDim.y;
  const int lane = threadIdx.x >> 4, j = threadIdx.x & 15;
  const int n = lane & 15, s = lane >> 4;
  const int kk = (j < 8) ? (8 * s + j) : (16 + 8 * s + (j - 8));
  const int row = t * 16 + n;
  const int col = kt * 32 + kk;
  float v = (col < Keff) ? W[(size_t)row * Keff + col] : 0.0f;
  out[(((size_t)t * nkt + kt) * 32 + lane) * 16 + j] = (_Float16)v;
}

// ---------------------------------------------------------------------------
// word embedding gather + pad-zero into info[N x 896] (f16). grid=N, block=128
// ---------------------------------------------------------------------------
__global__ __launch_bounds__(128) void word_kernel(const int* __restrict__ word_ids,
                                                   const float* __restrict__ word_emb,
                                                   _Float16* __restrict__ info) {
  const int i = blockIdx.x, tid = threadIdx.x;
  const int wid = word_ids[i];
  if (tid < 100)
    info[(size_t)i * IDIM + tid] = (_Float16)word_emb[(size_t)wid * 100 + tid];
  else if (tid < 128)
    info[(size_t)i * IDIM + IEFF + (tid - 100)] = (_Float16)0.0f;
}

// ---------------------------------------------------------------------------
// Fused char-LSTM (one direction per blockIdx.y) + masked avg/max/min pooling.
// block = 256 threads (8 wave32), 16 words per block.
// Wave w owns gate columns {g*128 + 16w : g=0..3} -> h columns [16w,16w+16).
// ---------------------------------------------------------------------------
__global__ __launch_bounds__(256) void lstm_kernel(
    const int* __restrict__ char_ids, const int* __restrict__ char_len,
    const float* __restrict__ char_emb,
    const _Float16* __restrict__ wihF, const _Float16* __restrict__ whhF,
    const float* __restrict__ bF,
    const _Float16* __restrict__ wihR, const _Float16* __restrict__ whhR,
    const float* __restrict__ bR,
    _Float16* __restrict__ info) {
  __shared__ __align__(16) _Float16 xbuf[256 * XSTR];  // 16 steps x 16 words x 64 (+pad)
  __shared__ __align__(16) _Float16 hbuf[16 * HSTR];   // 16 words x 128 (+pad)

  const int tid = threadIdx.x;
  const int lane = tid & 31;
  const int wv = tid >> 5;
  const int n = lane & 15, sh = lane >> 4;
  const int wbase = blockIdx.x * 16;
  const int dir = blockIdx.y;

  const _Float16* wih = dir ? wihR : wihF;
  const _Float16* whh = dir ? whhR : whhF;
  const float* bias = dir ? bR : bF;

  // ---- gather char embeddings (with reversal for dir==1) into LDS ----
  {
    const int word = tid >> 4, t = tid & 15;
    const int len = char_len[wbase + word];
    int st = t;
    if (dir) { st = len - 1 - t; if (st < 0) st = 0; }
    const int cid = char_ids[(wbase + word) * LSEQ + st];
    const float4* src = (const float4*)(char_emb + (size_t)cid * CDIM);
    _Float16* dst = xbuf + (t * 16 + word) * XSTR;
#pragma unroll
    for (int k8 = 0; k8 < 8; ++k8) {
      float4 a = src[2 * k8], b = src[2 * k8 + 1];
      v8h h8;
      h8[0] = (_Float16)a.x; h8[1] = (_Float16)a.y;
      h8[2] = (_Float16)a.z; h8[3] = (_Float16)a.w;
      h8[4] = (_Float16)b.x; h8[5] = (_Float16)b.y;
      h8[6] = (_Float16)b.z; h8[7] = (_Float16)b.w;
      *(v8h*)(dst + 8 * k8) = h8;
    }
  }
  for (int idx = tid; idx < 16 * HSTR; idx += 256) hbuf[idx] = (_Float16)0.0f;

  // per-lane bias for the 4 gate tiles this wave owns
  float bv[4];
#pragma unroll
  for (int g = 0; g < 4; ++g) bv[g] = bias[g * HDIM + 16 * wv + n];

  int ilen[8]; float flen[8];
#pragma unroll
  for (int e = 0; e < 8; ++e) {
    int r = char_len[wbase + e + 8 * sh];
    ilen[e] = r; flen[e] = (float)r;
  }

  v8f acc[4], sum, mx, mn;
  float cc[8];
#pragma unroll
  for (int e = 0; e < 8; ++e) {
    sum[e] = 0.0f; mx[e] = -1.0e9f; mn[e] = 1.0e9f; cc[e] = 0.0f;
    acc[0][e] = bv[0]; acc[1][e] = bv[1]; acc[2][e] = bv[2]; acc[3][e] = bv[3];
  }
  __syncthreads();

  for (int t = 0; t < LSEQ; ++t) {
    v16h xA[2], hA[4];
    {
      const _Float16* base = xbuf + t * 16 * XSTR;
#pragma unroll
      for (int kt = 0; kt < 2; ++kt) xA[kt] = loadA(base, XSTR, kt * 32, n, sh);
    }
#pragma unroll
    for (int kt = 0; kt < 4; ++kt) hA[kt] = loadA(hbuf, HSTR, kt * 32, n, sh);

#pragma unroll
    for (int g = 0; g < 4; ++g) {
      const int tile = g * 8 + wv;  // output cols [tile*16, tile*16+16)
#pragma unroll
      for (int kt = 0; kt < 2; ++kt) {
        v16h B = loadB(wih + (((size_t)tile * 2 + kt) * 32 + lane) * 16);
        acc[g] = wmma16(xA[kt], B, acc[g]);
      }
#pragma unroll
      for (int kt = 0; kt < 4; ++kt) {
        v16h B = loadB(whh + (((size_t)tile * 4 + kt) * 32 + lane) * 16);
        acc[g] = wmma16(hA[kt], B, acc[g]);
      }
    }
    __syncthreads();  // everyone done reading hbuf

#pragma unroll
    for (int e = 0; e < 8; ++e) {
      float iv = acc[0][e], fv = acc[1][e], gv = acc[2][e], ov = acc[3][e];
      float cn = sigm(fv) * cc[e] + sigm(iv) * fast_tanh(gv);
      cc[e] = cn;
      float hh = sigm(ov) * fast_tanh(cn);
      bool m = t < ilen[e];
      sum[e] += m ? hh : 0.0f;
      if (m) { mx[e] = fmaxf(mx[e], hh); mn[e] = fminf(mn[e], hh); }
      hbuf[(e + 8 * sh) * HSTR + 16 * wv + n] = (_Float16)hh;
      acc[0][e] = bv[0]; acc[1][e] = bv[1]; acc[2][e] = bv[2]; acc[3][e] = bv[3];
    }
    __syncthreads();  // new hbuf visible
  }

  // pooled features -> info: [avg_f avg_r max_f max_r min_f min_r] after word emb
  const int colbase = 100 + 128 * dir + 16 * wv + n;
#pragma unroll
  for (int e = 0; e < 8; ++e) {
    const size_t row = (size_t)(wbase + e + 8 * sh);
    _Float16* p = info + row * IDIM + colbase;
    p[0]   = (_Float16)(sum[e] / flen[e]);
    p[256] = (_Float16)mx[e];
    p[512] = (_Float16)mn[e];
  }
}

// ---------------------------------------------------------------------------
// out = tanh(info[N x 896] @ W_lin^T + b_lin), 16 rows / block, 8 waves,
// wave w owns 4 output tiles (cols 64w..64w+63). A staged in LDS.
// ---------------------------------------------------------------------------
__global__ __launch_bounds__(256) void out_gemm_kernel(
    const _Float16* __restrict__ info, const _Float16* __restrict__ wlin,
    const float* __restrict__ b_lin, float* __restrict__ out) {
  __shared__ __align__(16) _Float16 abuf[16 * ASTR];
  const int tid = threadIdx.x, lane = tid & 31, wv = tid >> 5;
  const int n = lane & 15, sh = lane >> 4;
  const int rowbase = blockIdx.x * 16;

  for (int idx = tid; idx < 16 * 112; idx += 256) {
    int r = idx / 112, c8 = idx % 112;
    *(v8h*)(abuf + r * ASTR + c8 * 8) =
        *(const v8h*)(info + (size_t)(rowbase + r) * IDIM + c8 * 8);
  }
  __syncthreads();

  v8f acc[4];
#pragma unroll
  for (int i = 0; i < 4; ++i) {
    float bvv = b_lin[(wv * 4 + i) * 16 + n];
#pragma unroll
    for (int e = 0; e < 8; ++e) acc[i][e] = bvv;
  }

  for (int kt = 0; kt < 28; ++kt) {
    v16h A = loadA(abuf, ASTR, kt * 32, n, sh);
#pragma unroll
    for (int i = 0; i < 4; ++i) {
      const int tile = wv * 4 + i;
      v16h B = loadB(wlin + (((size_t)tile * 28 + kt) * 32 + lane) * 16);
      acc[i] = wmma16(A, B, acc[i]);
    }
  }

#pragma unroll
  for (int i = 0; i < 4; ++i) {
    const int col0 = (wv * 4 + i) * 16;
#pragma unroll
    for (int e = 0; e < 8; ++e)
      out[(size_t)(rowbase + e + 8 * sh) * 512 + col0 + n] = fast_tanh(acc[i][e]);
  }
}

// ---------------------------------------------------------------------------
extern "C" void kernel_launch(void* const* d_in, const int* in_sizes, int n_in,
                              void* d_out, int out_size, void* d_ws, size_t ws_size,
                              hipStream_t stream) {
  (void)in_sizes; (void)n_in; (void)out_size; (void)ws_size;
  const int*   word_ids = (const int*)d_in[0];
  const int*   char_ids = (const int*)d_in[1];
  const int*   char_len = (const int*)d_in[2];
  const float* word_emb = (const float*)d_in[3];
  const float* char_emb = (const float*)d_in[4];
  const float* W_ih_f   = (const float*)d_in[5];
  const float* W_hh_f   = (const float*)d_in[6];
  const float* b_f      = (const float*)d_in[7];
  const float* W_ih_r   = (const float*)d_in[8];
  const float* W_hh_r   = (const float*)d_in[9];
  const float* b_r      = (const float*)d_in[10];
  const float* W_lin    = (const float*)d_in[11];
  const float* b_lin    = (const float*)d_in[12];
  float* out = (float*)d_out;

  _Float16* ws   = (_Float16*)d_ws;
  _Float16* wihF = ws + OFF_WIHF;
  _Float16* wihR = ws + OFF_WIHR;
  _Float16* whhF = ws + OFF_WHHF;
  _Float16* whhR = ws + OFF_WHHR;
  _Float16* wlin = ws + OFF_WLIN;
  _Float16* info = ws + OFF_INFO;

  // 1. pre-swizzle all weights into WMMA B-fragment layout (f16)
  swz_kernel<<<dim3(32, 2),  512, 0, stream>>>(W_ih_f, wihF, CDIM);
  swz_kernel<<<dim3(32, 2),  512, 0, stream>>>(W_ih_r, wihR, CDIM);
  swz_kernel<<<dim3(32, 4),  512, 0, stream>>>(W_hh_f, whhF, HDIM);
  swz_kernel<<<dim3(32, 4),  512, 0, stream>>>(W_hh_r, whhR, HDIM);
  swz_kernel<<<dim3(32, 28), 512, 0, stream>>>(W_lin,  wlin, IEFF);

  // 2. word embedding + pad zero into info
  word_kernel<<<NWORDS, 128, 0, stream>>>(word_ids, word_emb, info);

  // 3. bidirectional char LSTM + pooling -> info (y = direction)
  lstm_kernel<<<dim3(NWORDS / 16, 2), 256, 0, stream>>>(
      char_ids, char_len, char_emb, wihF, whhF, b_f, wihR, whhR, b_r, info);

  // 4. final linear + tanh
  out_gemm_kernel<<<NWORDS / 16, 256, 0, stream>>>(info, wlin, b_lin, out);
}
